// VQModel_61005715472985
// MI455X (gfx1250) — compile-verified, matrix operand order checked by reading, not run on hardware
//
#include <hip/hip_runtime.h>
#include <hip/hip_bf16.h>
#include <math.h>

// ---------------------------------------------------------------------------
// Types for CDNA5 WMMA (wave32): A/B = 16 bf16 per lane (8 VGPRs),
// C/D = 8 f32 per lane (8 VGPRs).
// ---------------------------------------------------------------------------
typedef __attribute__((ext_vector_type(16))) __bf16 v16bf;
typedef __attribute__((ext_vector_type(8)))  float  v8f;
typedef __attribute__((ext_vector_type(4)))  float  f32x4;

union FragBF {
    v16bf v;
    f32x4 h[2];   // two 16-byte halves (8 bf16 each)
};

__device__ __forceinline__ unsigned short f2bf(float f) {
    // round-to-nearest-even fp32 -> bf16
    unsigned u = __float_as_uint(f);
    unsigned r = (u + 0x7FFFu + ((u >> 16) & 1u)) >> 16;
    return (unsigned short)r;
}

// ---- CDNA5 async global->LDS copy (ASYNCcnt path), guarded ---------------
#if __has_builtin(__builtin_amdgcn_global_load_async_to_lds_b128)
#define USE_ASYNC_LDS 1
#else
#define USE_ASYNC_LDS 0
#endif

// builtin prototype (from hipcc diagnostic): param0 = AS(1) int4*, non-const
typedef int v4i __attribute__((vector_size(16)));
typedef __attribute__((address_space(1))) v4i* as1_v4i_p;
typedef __attribute__((address_space(3))) v4i* as3_v4i_p;

__device__ __forceinline__ void wait_async_lds() {
#if __has_builtin(__builtin_amdgcn_s_wait_asynccnt)
    __builtin_amdgcn_s_wait_asynccnt(0);
#endif
}

// ---------------------------------------------------------------------------
// Implicit-GEMM conv, k=4 s=2 p=1, all geometry compile-time.
//   MODE 0: forward conv        iy = 2*oy - 1 + ky
//   MODE 1: transposed conv, output-parity class pz = blockIdx.z:
//           oy = 2*oy2 + py ; ky = (1-py)+2*ty ; iy = (oy+1-ky)/2
// GEMM: M = 32*OH2*OW2, N = COUT, K = taps*CIN.
// Tile 128(M) x 64(N), K-chunk 32; block = 256 thr = 8 waves;
// wave w owns rows [16w,16w+16) x all 64 cols -> 4 accumulators,
// one shared A fragment per chunk -> 4 WMMA per stage.
// Weights pre-converted to bf16 [ (parity,) co ][ k ] (K-contiguous).
// ACT: 1 = relu, 2 = sigmoid.
// ---------------------------------------------------------------------------
template<int CIN, int COUT, int IH, int IW, int OH, int OW,
         int KK, int MODE, int ACT>
__global__ __launch_bounds__(256)
void conv_wmma(const float* __restrict__ in,
               const unsigned short* __restrict__ wbf,
               const float* __restrict__ bias,
               float* __restrict__ out)
{
    constexpr int OW2  = MODE ? (OW >> 1) : OW;
    constexpr int OH2  = MODE ? (OH >> 1) : OH;
    constexpr int OPIX = OH2 * OW2;            // power of two in all layers

    __align__(16) __shared__ unsigned short lA[128 * 32];
    __align__(16) __shared__ unsigned short lB[64 * 32];

    const int tid  = threadIdx.x;
    const int lane = tid & 31;
    const int wv   = tid >> 5;                 // wave 0..7 -> 16-row group
    const int tileM = blockIdx.x * 128;
    const int tileN = blockIdx.y * 64;
    const int pz = MODE ? blockIdx.z : 0;
    const int py = pz >> 1, px = pz & 1;

    v8f acc[4];
    #pragma unroll
    for (int g = 0; g < 4; ++g)
        acc[g] = (v8f){0.f,0.f,0.f,0.f,0.f,0.f,0.f,0.f};

    // ---- A staging assignment: 2 threads/row, 16 k-elements each ----------
    const int aRow = tid >> 1;                 // 0..127
    const int aK0  = (tid & 1) * 16;           // 0 / 16
    const int a_m    = tileM + aRow;
    const int a_nimg = a_m / OPIX;             // shift (OPIX = 2^n)
    const int a_p    = a_m & (OPIX - 1);
    int a_oy, a_ox;
    if (MODE) { int oy2 = a_p / OW2; int ox2 = a_p & (OW2 - 1);
                a_oy = oy2 * 2 + py;  a_ox = ox2 * 2 + px; }
    else      { a_oy = a_p / OW;      a_ox = a_p & (OW - 1); }

    auto loadA = [&](int k) -> float {
        if (k >= KK) return 0.f;
        const int kk = k / CIN;                // CIN const: shift or magic(3)
        const int ci = k - kk * CIN;
        int iy, ix; bool ok;
        if (MODE) {
            const int ty = kk >> 1, tx = kk & 1;
            const int ky = (1 - py) + 2 * ty;
            const int kx = (1 - px) + 2 * tx;
            const int iyn = a_oy + 1 - ky, ixn = a_ox + 1 - kx;
            iy = iyn >> 1; ix = ixn >> 1;
            ok = (iyn >= 0) && (ixn >= 0) && (iy < IH) && (ix < IW);
        } else {
            const int ky = kk >> 2, kx = kk & 3;
            iy = a_oy * 2 - 1 + ky;
            ix = a_ox * 2 - 1 + kx;
            ok = (iy >= 0) && (ix >= 0) && (iy < IH) && (ix < IW);
        }
        return ok ? in[(((long)a_nimg * CIN + ci) * IH + iy) * IW + ix] : 0.f;
    };

    // ---- B staging assignment: 4 threads/row, one 16B (8 bf16) each -------
    const int bRow = tid >> 2;                 // 0..63
    const int bK0  = (tid & 3) * 8;            // 0,8,16,24
    const bool bRowOk = (tileN + bRow) < COUT;
    const unsigned short* wRow =
        wbf + ((size_t)(MODE ? pz * COUT : 0) + (tileN + bRow)) * KK;

    unsigned int* lA32 = (unsigned int*)lA;

    for (int kc = 0; kc < KK; kc += 32) {
        // -------- A tile: gather fp32 activations, pack bf16 pairs --------
        #pragma unroll
        for (int j = 0; j < 16; j += 2) {
            const float v0 = loadA(kc + aK0 + j);
            const float v1 = loadA(kc + aK0 + j + 1);
            lA32[(aRow * 32 + aK0 + j) >> 1] =
                (unsigned)f2bf(v0) | ((unsigned)f2bf(v1) << 16);
        }
        // -------- B tile: contiguous bf16 weights, async -> LDS -----------
        {
            const bool ok = bRowOk && (kc + bK0) < KK;
#if USE_ASYNC_LDS
            if (ok) {
                __builtin_amdgcn_global_load_async_to_lds_b128(
                    (as1_v4i_p)(void*)(wRow + kc + bK0),
                    (as3_v4i_p)(void*)(&lB[bRow * 32 + bK0]),
                    0, 0);
            } else {
                *(f32x4*)(void*)&lB[bRow * 32 + bK0] = (f32x4){0.f,0.f,0.f,0.f};
            }
#else
            f32x4 wv4 = (f32x4){0.f,0.f,0.f,0.f};
            if (ok) wv4 = *(const f32x4*)(const void*)(wRow + kc + bK0);
            *(f32x4*)(void*)&lB[bRow * 32 + bK0] = wv4;
#endif
        }
#if USE_ASYNC_LDS
        wait_async_lds();
#endif
        __syncthreads();

        // -------- fragments per ISA 16-bit layout, 4x WMMA ----------------
        {
            const int r16 = lane & 15;
            const int kb  = (lane >> 4) * 8;   // lane-half K base
            FragBF fa;
            const f32x4* As = (const f32x4*)(const void*)
                              &lA[(wv * 16 + r16) * 32 + kb];
            fa.h[0] = As[0];                   // K = kb .. kb+7
            fa.h[1] = As[2];                   // K = kb+16 .. kb+23
            #pragma unroll
            for (int g = 0; g < 4; ++g) {
                FragBF fb;
                const f32x4* Bs = (const f32x4*)(const void*)
                                  &lB[(g * 16 + r16) * 32 + kb];
                fb.h[0] = Bs[0];  fb.h[1] = Bs[2];
                acc[g] = __builtin_amdgcn_wmma_f32_16x16x32_bf16(
                             false, fa.v, false, fb.v, (short)0, acc[g],
                             false, false);
            }
        }
        __syncthreads();
    }

    // ---- epilogue: bias + activation + NCHW store (all shifts) -----------
    const int r16  = lane & 15;
    const int half = lane >> 4;
    #pragma unroll
    for (int g = 0; g < 4; ++g) {
        const int n = tileN + g * 16 + r16;
        if (n < COUT) {
            const float bv = bias[n];
            #pragma unroll
            for (int v = 0; v < 8; ++v) {
                const int m    = tileM + wv * 16 + v + half * 8;
                const int nimg = m / OPIX;
                const int p    = m & (OPIX - 1);
                int oy, ox;
                if (MODE) { int oy2 = p / OW2; int ox2 = p & (OW2 - 1);
                            oy = oy2 * 2 + py; ox = ox2 * 2 + px; }
                else      { oy = p / OW;       ox = p & (OW - 1); }
                float r = acc[g][v] + bv;
                if (ACT == 1)      r = fmaxf(r, 0.f);
                else if (ACT == 2) r = 1.f / (1.f + __expf(-r));
                out[(((long)nimg * COUT + n) * OH + oy) * OW + ox] = r;
            }
        }
    }
}

// ---------------------------------------------------------------------------
// Weight prep: fp32 HWIO -> bf16 [co][K]  (K = (ky*4+kx)*Cin+ci contiguous)
// ---------------------------------------------------------------------------
__global__ void prep_w_conv(const float* __restrict__ w,
                            unsigned short* __restrict__ wbf,
                            int K, int Cout)
{
    const int i = blockIdx.x * 256 + threadIdx.x;
    if (i < K * Cout) {
        const int co = i / K;
        const int k  = i - co * K;
        wbf[i] = f2bf(w[(long)k * Cout + co]);
    }
}

// fp32 HWIO -> bf16 [parity][co][4*Cin] for transposed conv
__global__ void prep_w_convT(const float* __restrict__ w,
                             unsigned short* __restrict__ wbf,
                             int Cin, int Cout)
{
    const int K4 = 4 * Cin;
    const int i = blockIdx.x * 256 + threadIdx.x;
    if (i < 4 * Cout * K4) {
        const int pz = i / (Cout * K4);
        const int r  = i - pz * (Cout * K4);
        const int co = r / K4;
        const int k  = r - co * K4;
        const int t  = k / Cin;
        const int ci = k - t * Cin;
        const int py = pz >> 1, px = pz & 1;
        const int ty = t >> 1,  tx = t & 1;
        const int ky = (1 - py) + 2 * ty;
        const int kx = (1 - px) + 2 * tx;
        wbf[i] = f2bf(w[((long)((ky * 4 + kx) * Cin + ci)) * Cout + co]);
    }
}

// ---------------------------------------------------------------------------
// Vector quantize: one wave per latent vector (dim 64 -> 2 dims per lane).
// z is NCHW (32,64,16,16); codebook (512,64) row-major.
// ---------------------------------------------------------------------------
__global__ __launch_bounds__(256)
void vq_kernel(const float* __restrict__ z, const float* __restrict__ cb,
               float* __restrict__ q, int* __restrict__ hist,
               float* __restrict__ sse, int NV)
{
    const int lane = threadIdx.x & 31;
    const int wv   = threadIdx.x >> 5;
    const int vec  = blockIdx.x * 8 + wv;
    if (vec >= NV) return;

    const int n   = vec >> 8;
    const int rem = vec & 255;
    const int y   = rem >> 4, x = rem & 15;
    const long zbase = ((long)n * 64) * 256 + y * 16 + x;  // channel stride 256

    const float z0 = z[zbase + (2 * lane)     * 256];
    const float z1 = z[zbase + (2 * lane + 1) * 256];

    float best = 3.4e38f; int bestc = 0;
    for (int c = 0; c < 512; ++c) {
        const float* cr = cb + c * 64 + 2 * lane;
        const float d0 = z0 - cr[0];
        const float d1 = z1 - cr[1];
        float part = d0 * d0 + d1 * d1;
        #pragma unroll
        for (int off = 16; off > 0; off >>= 1)
            part += __shfl_xor(part, off, 32);
        if (part < best) { best = part; bestc = c; }
    }

    const float q0 = cb[bestc * 64 + 2 * lane];
    const float q1 = cb[bestc * 64 + 2 * lane + 1];
    q[zbase + (2 * lane)     * 256] = q0;
    q[zbase + (2 * lane + 1) * 256] = q1;

    const float e0 = z0 - q0, e1 = z1 - q1;
    float part = e0 * e0 + e1 * e1;
    #pragma unroll
    for (int off = 16; off > 0; off >>= 1)
        part += __shfl_xor(part, off, 32);
    if (lane == 0) {
        atomicAdd(sse, part);
        atomicAdd(&hist[bestc], 1);
    }
}

__global__ void vq_finalize(const int* __restrict__ hist,
                            const float* __restrict__ sse,
                            float* __restrict__ out3)
{
    __shared__ int cnt;
    if (threadIdx.x == 0) cnt = 0;
    __syncthreads();
    if (threadIdx.x < 512 && hist[threadIdx.x] > 0) atomicAdd(&cnt, 1);
    __syncthreads();
    if (threadIdx.x == 0) {
        const float m = *sse / (8192.0f * 64.0f);   // mean over 32*64*16*16
        out3[0] = m;            // commitment
        out3[1] = m;            // codebook loss (same value)
        out3[2] = (float)cnt;   // unique codes
    }
}

// ---------------------------------------------------------------------------
extern "C" void kernel_launch(void* const* d_in, const int* in_sizes, int n_in,
                              void* d_out, int out_size, void* d_ws, size_t ws_size,
                              hipStream_t stream)
{
    (void)in_sizes; (void)n_in; (void)ws_size; (void)out_size;
    const float* x    = (const float*)d_in[0];
    const float* ew1  = (const float*)d_in[1];
    const float* eb1  = (const float*)d_in[2];
    const float* ew2  = (const float*)d_in[3];
    const float* eb2  = (const float*)d_in[4];
    const float* ew3  = (const float*)d_in[5];
    const float* eb3  = (const float*)d_in[6];
    const float* cbk  = (const float*)d_in[7];
    const float* dw1  = (const float*)d_in[8];
    const float* db1  = (const float*)d_in[9];
    const float* dw2  = (const float*)d_in[10];
    const float* db2  = (const float*)d_in[11];
    const float* dw3  = (const float*)d_in[12];
    const float* db3  = (const float*)d_in[13];
    float* out = (float*)d_out;

    char* ws = (char*)d_ws;
    size_t off = 0;
    float* bufA = (float*)(ws + off); off += 134217728;        // h1 / d2
    float* bufB = (float*)(ws + off); off += 67108864;         // h2 / d1
    float* z    = (float*)(ws + off); off += 2097152;
    float* q    = (float*)(ws + off); off += 2097152;
    int*   hist = (int*)  (ws + off); off += 2048;
    float* sse  = (float*)(ws + off); off += 64;
    unsigned short* wbf1 = (unsigned short*)(ws + off); off += 2u*256*48;
    unsigned short* wbf2 = (unsigned short*)(ws + off); off += 2u*512*4096;
    unsigned short* wbf3 = (unsigned short*)(ws + off); off += 2u*64*8192;
    unsigned short* wtb1 = (unsigned short*)(ws + off); off += 2u*4*512*256;
    unsigned short* wtb2 = (unsigned short*)(ws + off); off += 2u*4*256*2048;
    unsigned short* wtb3 = (unsigned short*)(ws + off); off += 2u*4*3*1024;

    (void)hipMemsetAsync(hist, 0, 512 * sizeof(int) + sizeof(float), stream);

    dim3 blk(256);
    // ---- weight prep (fp32 HWIO -> bf16 K-contiguous) ---------------------
    prep_w_conv <<<dim3((256*48   + 255)/256), blk, 0, stream>>>(ew1, wbf1,   48, 256);
    prep_w_conv <<<dim3((512*4096 + 255)/256), blk, 0, stream>>>(ew2, wbf2, 4096, 512);
    prep_w_conv <<<dim3((64*8192  + 255)/256), blk, 0, stream>>>(ew3, wbf3, 8192,  64);
    prep_w_convT<<<dim3((4*512*256  + 255)/256), blk, 0, stream>>>(dw1, wtb1,  64, 512);
    prep_w_convT<<<dim3((4*256*2048 + 255)/256), blk, 0, stream>>>(dw2, wtb2, 512, 256);
    prep_w_convT<<<dim3((4*3*1024   + 255)/256), blk, 0, stream>>>(dw3, wtb3, 256,   3);

    // ---- encoder (MODE 0): M = 32*OH*OW, K = 16*Cin -----------------------
    conv_wmma<  3,256,128,128, 64, 64,  48,0,1><<<dim3(1024,4,1), blk, 0, stream>>>(x,    wbf1, eb1, bufA);
    conv_wmma<256,512, 64, 64, 32, 32,4096,0,1><<<dim3( 256,8,1), blk, 0, stream>>>(bufA, wbf2, eb2, bufB);
    conv_wmma<512, 64, 32, 32, 16, 16,8192,0,1><<<dim3(  64,1,1), blk, 0, stream>>>(bufB, wbf3, eb3, z);

    // ---- vector quantize (straight-through: decoder consumes q) -----------
    vq_kernel<<<dim3(8192/8), blk, 0, stream>>>(z, cbk, q, hist, sse, 8192);
    vq_finalize<<<1, 512, 0, stream>>>(hist, sse, out + 1572864);

    // ---- decoder (MODE 1, 4 parity classes): M = 32*(OH/2)*(OW/2), K=4*Cin
    conv_wmma< 64,512, 16, 16, 32, 32, 256,1,1><<<dim3(  64,8,4), blk, 0, stream>>>(q,    wtb1, db1, bufB);
    conv_wmma<512,256, 32, 32, 64, 64,2048,1,1><<<dim3( 256,4,4), blk, 0, stream>>>(bufB, wtb2, db2, bufA);
    conv_wmma<256,  3, 64, 64,128,128,1024,1,2><<<dim3(1024,1,4), blk, 0, stream>>>(bufA, wtb3, db3, out);
}
